// Reflect_38482906972800
// MI455X (gfx1250) — compile-verified, matrix operand order checked by reading, not run on hardware
//
#include <hip/hip_runtime.h>

// ---------------------------------------------------------------------------
// Attention forward, B=4, S=2048, D=1024 (fp32 in/out), gfx1250 (MI455X).
// All GEMMs via v_wmma_f32_16x16x32_bf16 (bf16 operands, f32 accumulate).
// Wave tile 32x64 (2 A-frags x 4 B-frags -> 8 accumulators), block tile
// 128x128 (8 waves). Fragment loads hoisted per K-chunk so the compiler can
// clause 12x global_load_b128 and issue 8 WMMAs back-to-back.
// Workspace layout (bytes):
//   [0,6MB)     wT : 3x [1024][1024] bf16, W transposed
//   [6,22MB)    xb : [8192][1024] bf16 (x converted once)
//   [22,38MB)   q  : [8192][1024] bf16 row-major
//   [38,54MB)   k  : [8192][1024] bf16 row-major
//   [54,70MB)   vT : [4][1024][2048] bf16 (V transposed per batch)
//   [70,134MB)  y  : [8192][2048] f32 scores; softmax overwrites each row's
//                    first half in place with [2048] bf16 probabilities
// ---------------------------------------------------------------------------

typedef __attribute__((ext_vector_type(16))) __bf16 v16bf;
typedef __attribute__((ext_vector_type(8)))  float  v8f;

#define NB 4
#define SS 2048
#define DD 1024
#define BS 8192
#define SCALE 0.03125f

static __device__ __forceinline__ unsigned short f2bfu(float f) {
  union { float f; unsigned u; } c; c.f = f;
  unsigned r = c.u + 0x7FFFu + ((c.u >> 16) & 1u);   // round-to-nearest-even
  return (unsigned short)(r >> 16);
}
static __device__ __forceinline__ unsigned pk2(float a, float b) {
  return (unsigned)f2bfu(a) | ((unsigned)f2bfu(b) << 16);
}

// bf16 A/B fragment: lane holds 16 elems of row `row` (stride ld, ushorts) at
// K offsets [c0..c0+7] and [c0+16..c0+23], where c0 = kChunk + 8*(lane>=16).
static __device__ __forceinline__ v16bf ldfrag(const unsigned short* __restrict__ base,
                                               int ld, int row, int c0) {
  const unsigned short* p = base + (size_t)row * ld + c0;
  uint4 a = *(const uint4*)p;
  uint4 b = *(const uint4*)(p + 16);
  union { unsigned u[8]; v16bf v; } pk;
  pk.u[0]=a.x; pk.u[1]=a.y; pk.u[2]=a.z; pk.u[3]=a.w;
  pk.u[4]=b.x; pk.u[5]=b.y; pk.u[6]=b.z; pk.u[7]=b.w;
  return pk.v;
}

static __device__ __forceinline__ v8f wmma_bf16(v16bf a, v16bf b, v8f c) {
  return __builtin_amdgcn_wmma_f32_16x16x32_bf16(false, a, false, b,
                                                 (short)0, c, false, false);
}

// --------------------------- kernel 0a: weights -> wT bf16 ------------------
__global__ void k_cvt_weights(const float* __restrict__ qw,
                              const float* __restrict__ kw,
                              const float* __restrict__ vw,
                              unsigned short* __restrict__ wT) {
  int idx = blockIdx.x * 256 + threadIdx.x;
  if (idx >= 3 * DD * DD) return;
  int w = idx >> 20, r = idx & 0xFFFFF;
  int k = r >> 10, n = r & 1023;                 // coalesced read over n
  const float* src = (w == 0) ? qw : (w == 1) ? kw : vw;
  wT[((size_t)w << 20) + ((size_t)n << 10) + k] = f2bfu(src[((size_t)k << 10) + n]);
}

// --------------------------- kernel 0b: x -> xb bf16 ------------------------
__global__ void k_cvt_x(const float* __restrict__ x,
                        unsigned short* __restrict__ xb) {
  int idx = blockIdx.x * 256 + threadIdx.x;      // one float4 per thread
  float4 v = ((const float4*)x)[idx];
  union { unsigned u[2]; uint2 q; } pk;
  pk.u[0] = pk2(v.x, v.y);
  pk.u[1] = pk2(v.z, v.w);
  ((uint2*)xb)[idx] = pk.q;
}

// --------------------------- kernel 1: projections --------------------------
// grid: (BS/128, DD/128, 3). z: 0->Q(row-major) 1->K(row-major) 2->V(transposed)
__global__ void k_proj(const unsigned short* __restrict__ xb,
                       const unsigned short* __restrict__ wTall,
                       unsigned short* __restrict__ qo,
                       unsigned short* __restrict__ ko,
                       unsigned short* __restrict__ vT) {
  const int mode = blockIdx.z;
  const unsigned short* wT = wTall + ((size_t)mode << 20);
  const int lane = threadIdx.x & 31, wv = threadIdx.x >> 5;
  const int g = lane >> 4, nl = lane & 15;
  const int mbase = blockIdx.x * 128 + (wv & 3) * 32;
  const int nbase = blockIdx.y * 128 + (wv >> 2) * 64;
  const int ar0 = mbase + nl, ar1 = mbase + 16 + nl;

  v8f acc[8];
  #pragma unroll
  for (int t = 0; t < 8; ++t) acc[t] = (v8f){0,0,0,0,0,0,0,0};

  #pragma unroll 1
  for (int kc = 0; kc < DD; kc += 32) {
    const int c0 = kc + g * 8;
    v16bf a0 = ldfrag(xb, DD, ar0, c0);
    v16bf a1 = ldfrag(xb, DD, ar1, c0);
    v16bf b0 = ldfrag(wT, DD, nbase +  0 + nl, c0);
    v16bf b1 = ldfrag(wT, DD, nbase + 16 + nl, c0);
    v16bf b2 = ldfrag(wT, DD, nbase + 32 + nl, c0);
    v16bf b3 = ldfrag(wT, DD, nbase + 48 + nl, c0);
    __builtin_prefetch(xb + (size_t)ar0 * DD + c0 + 64, 0, 3);
    acc[0] = wmma_bf16(a0, b0, acc[0]);
    acc[1] = wmma_bf16(a0, b1, acc[1]);
    acc[2] = wmma_bf16(a0, b2, acc[2]);
    acc[3] = wmma_bf16(a0, b3, acc[3]);
    acc[4] = wmma_bf16(a1, b0, acc[4]);
    acc[5] = wmma_bf16(a1, b1, acc[5]);
    acc[6] = wmma_bf16(a1, b2, acc[6]);
    acc[7] = wmma_bf16(a1, b3, acc[7]);
  }

  if (mode < 2) {                         // Q / K: row-major bf16
    unsigned short* out = (mode == 0) ? qo : ko;
    #pragma unroll
    for (int i = 0; i < 2; ++i)
      #pragma unroll
      for (int j = 0; j < 4; ++j) {
        const int col = nbase + j * 16 + nl;
        #pragma unroll
        for (int r = 0; r < 8; ++r) {
          int m = mbase + i * 16 + r + 8 * g;
          out[(size_t)m * DD + col] = f2bfu(acc[i * 4 + j][r]);
        }
      }
  } else {                                // V transposed: vT[b][d][s]
    const int b = mbase >> 11;
    #pragma unroll
    for (int i = 0; i < 2; ++i) {
      const int s0 = (mbase & (SS - 1)) + i * 16 + 8 * g;
      #pragma unroll
      for (int j = 0; j < 4; ++j) {
        const int d = nbase + j * 16 + nl;
        v8f a = acc[i * 4 + j];
        union { unsigned u[4]; uint4 q; } pk;
        pk.u[0] = pk2(a[0], a[1]);
        pk.u[1] = pk2(a[2], a[3]);
        pk.u[2] = pk2(a[4], a[5]);
        pk.u[3] = pk2(a[6], a[7]);
        *(uint4*)(vT + ((size_t)(b * DD + d)) * SS + s0) = pk.q;
      }
    }
  }
}

// --------------------------- kernel 2: scores = q k^T * scale + mask --------
// grid: (SS/128, SS/128, NB)
__global__ void k_scores(const unsigned short* __restrict__ q,
                         const unsigned short* __restrict__ k,
                         const int* __restrict__ mask,
                         float* __restrict__ y) {
  const int b = blockIdx.z;
  const int lane = threadIdx.x & 31, wv = threadIdx.x >> 5;
  const int g = lane >> 4, nl = lane & 15;
  const int sbase = blockIdx.x * 128 + (wv & 3) * 32;
  const int zbase = blockIdx.y * 128 + (wv >> 2) * 64;
  const int ar0 = b * SS + sbase + nl, ar1 = ar0 + 16;

  v8f acc[8];
  #pragma unroll
  for (int t = 0; t < 8; ++t) acc[t] = (v8f){0,0,0,0,0,0,0,0};

  #pragma unroll 1
  for (int dc = 0; dc < DD; dc += 32) {
    const int c0 = dc + g * 8;
    v16bf a0 = ldfrag(q, DD, ar0, c0);
    v16bf a1 = ldfrag(q, DD, ar1, c0);
    v16bf b0 = ldfrag(k, DD, b * SS + zbase +  0 + nl, c0);
    v16bf b1 = ldfrag(k, DD, b * SS + zbase + 16 + nl, c0);
    v16bf b2 = ldfrag(k, DD, b * SS + zbase + 32 + nl, c0);
    v16bf b3 = ldfrag(k, DD, b * SS + zbase + 48 + nl, c0);
    __builtin_prefetch(q + (size_t)ar0 * DD + c0 + 64, 0, 3);
    acc[0] = wmma_bf16(a0, b0, acc[0]);
    acc[1] = wmma_bf16(a0, b1, acc[1]);
    acc[2] = wmma_bf16(a0, b2, acc[2]);
    acc[3] = wmma_bf16(a0, b3, acc[3]);
    acc[4] = wmma_bf16(a1, b0, acc[4]);
    acc[5] = wmma_bf16(a1, b1, acc[5]);
    acc[6] = wmma_bf16(a1, b2, acc[6]);
    acc[7] = wmma_bf16(a1, b3, acc[7]);
  }

  float madd[2][8];
  #pragma unroll
  for (int i = 0; i < 2; ++i)
    #pragma unroll
    for (int r = 0; r < 8; ++r) {
      int s = sbase + i * 16 + r + 8 * g;
      madd[i][r] = mask[b * SS + s] ? 0.0f : -1e9f;   // additive query-row mask
    }
  #pragma unroll
  for (int i = 0; i < 2; ++i)
    #pragma unroll
    for (int j = 0; j < 4; ++j) {
      const int z = zbase + j * 16 + nl;
      #pragma unroll
      for (int r = 0; r < 8; ++r) {
        int s = sbase + i * 16 + r + 8 * g;
        y[((size_t)(b * SS + s)) * SS + z] = acc[i * 4 + j][r] * SCALE + madd[i][r];
      }
    }
}

// --------------------------- kernel 3: softmax row, P -> bf16 in place ------
__global__ void k_softmax(float* __restrict__ y) {
  __shared__ float red[256];
  const int t = threadIdx.x;
  float* rp = y + (size_t)blockIdx.x * SS;
  float4 v0 = *(const float4*)(rp + t * 8);
  float4 v1 = *(const float4*)(rp + t * 8 + 4);

  float m = fmaxf(fmaxf(fmaxf(v0.x, v0.y), fmaxf(v0.z, v0.w)),
                  fmaxf(fmaxf(v1.x, v1.y), fmaxf(v1.z, v1.w)));
  red[t] = m; __syncthreads();
  for (int s = 128; s > 0; s >>= 1) {
    if (t < s) red[t] = fmaxf(red[t], red[t + s]);
    __syncthreads();
  }
  const float rmax = red[0]; __syncthreads();

  float e[8];
  e[0]=expf(v0.x-rmax); e[1]=expf(v0.y-rmax); e[2]=expf(v0.z-rmax); e[3]=expf(v0.w-rmax);
  e[4]=expf(v1.x-rmax); e[5]=expf(v1.y-rmax); e[6]=expf(v1.z-rmax); e[7]=expf(v1.w-rmax);
  float sum = (e[0]+e[1])+(e[2]+e[3])+((e[4]+e[5])+(e[6]+e[7]));
  red[t] = sum; __syncthreads();
  for (int s = 128; s > 0; s >>= 1) {
    if (t < s) red[t] += red[t + s];
    __syncthreads();
  }
  const float inv = 1.0f / red[0];
  __syncthreads();

  union { unsigned u[4]; uint4 q; } pk;    // bf16 P over first half of the row
  pk.u[0] = pk2(e[0]*inv, e[1]*inv);
  pk.u[1] = pk2(e[2]*inv, e[3]*inv);
  pk.u[2] = pk2(e[4]*inv, e[5]*inv);
  pk.u[3] = pk2(e[6]*inv, e[7]*inv);
  ((uint4*)rp)[t] = pk.q;                  // ushort index t*8 == byte t*16
}

// --------------------------- kernel 4: out = P @ V ---------------------------
// grid: (BS/128, DD/128). P rows: bf16[2048] at stride 4096 ushorts inside y.
__global__ void k_pv(const float* __restrict__ y,
                     const unsigned short* __restrict__ vT,
                     float* __restrict__ out) {
  const unsigned short* P = (const unsigned short*)y;
  const int lane = threadIdx.x & 31, wv = threadIdx.x >> 5;
  const int g = lane >> 4, nl = lane & 15;
  const int mbase = blockIdx.x * 128 + (wv & 3) * 32;
  const int dbase = blockIdx.y * 128 + (wv >> 2) * 64;
  const int b = mbase >> 11;
  const int ar0 = mbase + nl, ar1 = ar0 + 16;

  v8f acc[8];
  #pragma unroll
  for (int t = 0; t < 8; ++t) acc[t] = (v8f){0,0,0,0,0,0,0,0};

  #pragma unroll 1
  for (int zc = 0; zc < SS; zc += 32) {
    const int c0 = zc + g * 8;
    v16bf a0 = ldfrag(P, 2 * SS, ar0, c0);           // row stride 4096 ushorts
    v16bf a1 = ldfrag(P, 2 * SS, ar1, c0);
    v16bf b0 = ldfrag(vT, SS, b * DD + dbase +  0 + nl, c0);
    v16bf b1 = ldfrag(vT, SS, b * DD + dbase + 16 + nl, c0);
    v16bf b2 = ldfrag(vT, SS, b * DD + dbase + 32 + nl, c0);
    v16bf b3 = ldfrag(vT, SS, b * DD + dbase + 48 + nl, c0);
    acc[0] = wmma_bf16(a0, b0, acc[0]);
    acc[1] = wmma_bf16(a0, b1, acc[1]);
    acc[2] = wmma_bf16(a0, b2, acc[2]);
    acc[3] = wmma_bf16(a0, b3, acc[3]);
    acc[4] = wmma_bf16(a1, b0, acc[4]);
    acc[5] = wmma_bf16(a1, b1, acc[5]);
    acc[6] = wmma_bf16(a1, b2, acc[6]);
    acc[7] = wmma_bf16(a1, b3, acc[7]);
  }

  #pragma unroll
  for (int i = 0; i < 2; ++i)
    #pragma unroll
    for (int j = 0; j < 4; ++j) {
      const int d = dbase + j * 16 + nl;
      #pragma unroll
      for (int r = 0; r < 8; ++r) {
        int m = mbase + i * 16 + r + 8 * g;
        out[(size_t)m * DD + d] = acc[i * 4 + j][r];
      }
    }
}

// ---------------------------------------------------------------------------
extern "C" void kernel_launch(void* const* d_in, const int* in_sizes, int n_in,
                              void* d_out, int out_size, void* d_ws, size_t ws_size,
                              hipStream_t stream) {
  const float* x    = (const float*)d_in[0];
  const int*   mask = (const int*)  d_in[1];
  const float* qw   = (const float*)d_in[2];
  const float* kw   = (const float*)d_in[3];
  const float* vw   = (const float*)d_in[4];
  float* out = (float*)d_out;

  char* ws = (char*)d_ws;
  unsigned short* wT = (unsigned short*)(ws);                          //  6 MB
  unsigned short* xb = (unsigned short*)(ws + (size_t)6  * (1 << 20)); // 16 MB
  unsigned short* q  = (unsigned short*)(ws + (size_t)22 * (1 << 20)); // 16 MB
  unsigned short* k  = (unsigned short*)(ws + (size_t)38 * (1 << 20)); // 16 MB
  unsigned short* vT = (unsigned short*)(ws + (size_t)54 * (1 << 20)); // 16 MB
  float*          y  = (float*)         (ws + (size_t)70 * (1 << 20)); // 64 MB

  k_cvt_weights<<<(3 * DD * DD) / 256, 256, 0, stream>>>(qw, kw, vw, wT);
  k_cvt_x<<<(BS * DD / 4) / 256, 256, 0, stream>>>(x, xb);
  k_proj<<<dim3(BS / 128, DD / 128, 3), 256, 0, stream>>>(xb, wT, q, k, vT);
  k_scores<<<dim3(SS / 128, SS / 128, NB), 256, 0, stream>>>(q, k, mask, y);
  k_softmax<<<BS, 256, 0, stream>>>(y);
  k_pv<<<dim3(BS / 128, DD / 128), 256, 0, stream>>>(y, vT, out);
}
